// KalmanConvLSTMHead_62397284876569
// MI455X (gfx1250) — matrix operand order repaired and checked by reading, not compiled, not run on hardware
//
#include <hip/hip_runtime.h>
#include <hip/hip_bf16.h>

#define DEVI static __device__ __forceinline__

typedef __attribute__((ext_vector_type(16))) __bf16 v16bf;
typedef __attribute__((ext_vector_type(8)))  float  v8f;
typedef __attribute__((ext_vector_type(16))) unsigned short v16u;
typedef __attribute__((ext_vector_type(8)))  unsigned short v8u;

union BFPack { v16u u; v16bf b; };

constexpr int Bn   = 8;
constexpr int Sn   = 16;
constexpr int CINn = 128;
constexpr int Hn   = 64;
constexpr int Wn   = 64;
constexpr int HWn  = Hn * Wn;        // 4096
constexpr int BSn  = Bn * Sn;        // 128
constexpr int NPIX = Bn * HWn;       // 32768

DEVI unsigned short f2bf(float f) {
  unsigned int u = __float_as_uint(f);
  u += 0x7FFFu + ((u >> 16) & 1u);
  return (unsigned short)(u >> 16);
}

DEVI float sigm(float x) { return 1.f / (1.f + expf(-x)); }

DEVI v8f wmma_bf16(v16bf a, v16bf b, v8f c) {
  return __builtin_amdgcn_wmma_f32_16x16x32_bf16(false, a, false, b, (short)0, c,
                                                 false, false);
}

// A-fragment (16x32) directly from a bf16 channel-contiguous activation row.
// Lanes 0-15 hold K={0..7,16..23}, lanes 16-31 hold K={8..15,24..31}:
// two aligned 16B vector loads, no conversion ALU.
DEVI v16bf load_a_bf16(const unsigned short* src, bool valid) {
  BFPack p;
  if (valid) {
    int kb = ((threadIdx.x >> 4) & 1) * 8;
    v8u lo = *(const v8u*)(src + kb);
    v8u hi = *(const v8u*)(src + 16 + kb);
    p.u = __builtin_shufflevector(lo, hi, 0, 1, 2, 3, 4, 5, 6, 7,
                                  8, 9, 10, 11, 12, 13, 14, 15);
  } else {
    p.u = (v16u){0, 0, 0, 0, 0, 0, 0, 0, 0, 0, 0, 0, 0, 0, 0, 0};
  }
  return p.b;
}

// B-fragment from pre-packed lane-major buffer: lane reads 16 contiguous u16.
DEVI v16bf load_b_packed(const unsigned short* pk) {
  int lane = threadIdx.x & 31;
  BFPack p;
  p.u = *(const v16u*)(pk + lane * 16);
  return p.b;
}

// ---------------------------------------------------------------- utilities
__global__ void zero_kernel(float* p, long long n) {
  long long i = (long long)blockIdx.x * blockDim.x + threadIdx.x;
  if (i < n) p[i] = 0.f;
}

__global__ void init_P_kernel(const float* __restrict__ P_init, float* __restrict__ P) {
  int i = blockIdx.x * blockDim.x + threadIdx.x;
  if (i < Bn * 64 * 64) P[i] = P_init[i & 4095];
}

// ------------------------------------------------------ weight pre-packing
// Pack B-fragments: within a 512-element tile, index = lane*16 + h with
// n_local = lane&15, k_local = ((lane>>4)&1)*8 + h + (h >= 8 ? 8 : 0).
__global__ void pack_weights_kernel(const float* __restrict__ clstm_w,
                                    const float* __restrict__ fcf_w,
                                    const float* __restrict__ proj_w,
                                    unsigned short* __restrict__ pkC,
                                    unsigned short* __restrict__ pkF,
                                    unsigned short* __restrict__ pkP) {
  const int NC = 16 * 9 * 4 * 512;   // 294912
  const int NF = 4 * 2 * 512;        // 4096
  const int NP = 4 * 4 * 512;        // 8192
  int idx = blockIdx.x * blockDim.x + threadIdx.x;
  if (idx < NC) {
    int e = idx & 511, tI = idx >> 9;
    int kc  = tI & 3;
    int tap = (tI >> 2) % 9;
    int nt  = (tI >> 2) / 9;
    int lane = e >> 4, h = e & 15;
    int n = nt * 16 + (lane & 15);                       // out channel 0..255
    int k = kc * 32 + ((lane >> 4) & 1) * 8 + h + (h >= 8 ? 8 : 0);  // in channel 0..127
    int dy = tap / 3, dx = tap % 3;
    pkC[idx] = f2bf(clstm_w[((n * 128 + k) * 3 + dy) * 3 + dx]);
  } else if (idx < NC + NF) {
    int id2 = idx - NC;
    int e = id2 & 511, tI = id2 >> 9;
    int kc = tI & 1, nt = tI >> 1;
    int lane = e >> 4, h = e & 15;
    int n = nt * 16 + (lane & 15);
    int k = kc * 32 + ((lane >> 4) & 1) * 8 + h + (h >= 8 ? 8 : 0);
    pkF[id2] = f2bf(fcf_w[n * 64 + k]);
  } else if (idx < NC + NF + NP) {
    int id3 = idx - NC - NF;
    int e = id3 & 511, tI = id3 >> 9;
    int kc = tI & 3, nt = tI >> 2;
    int lane = e >> 4, h = e & 15;
    int n = nt * 16 + (lane & 15);
    int k = kc * 32 + ((lane >> 4) & 1) * 8 + h + (h >= 8 ? 8 : 0);
    pkP[id3] = f2bf(proj_w[n * 128 + k]);
  }
}

// ---------------------------------------------------- 1x1 projection + stats
// grid: 128 images * 32 blocks, 256 threads (8 waves, 16 pixels each)
__global__ void __launch_bounds__(256)
proj_gemm_kernel(const float* __restrict__ x, const unsigned short* __restrict__ pkP,
                 float* __restrict__ x0raw, float* __restrict__ chansum,
                 float* __restrict__ chansumsq, float* __restrict__ ymean_raw) {
  __shared__ float csum[64], csum2[64];
  int tid = threadIdx.x;
  if (tid < 64) { csum[tid] = 0.f; csum2[tid] = 0.f; }
  __syncthreads();

  int img = blockIdx.x >> 5;     // 0..127 (= b*S + s)
  int blk = blockIdx.x & 31;
  int wave = tid >> 5, lane = tid & 31;
  int m = lane & 15;
  int pix0 = blk * 128 + wave * 16;
  int pix  = pix0 + m;
  const float* xin = x + (size_t)img * CINn * HWn;   // [c][p]

  v8f z = {0.f, 0.f, 0.f, 0.f, 0.f, 0.f, 0.f, 0.f};
  v8f acc[4];
#pragma unroll
  for (int nt = 0; nt < 4; ++nt) acc[nt] = z;

#pragma unroll
  for (int kc = 0; kc < 4; ++kc) {
    // strided A gather (channel stride = HW), fp32 -> bf16 once per element
    BFPack pa;
    int kb = ((lane >> 4) & 1) * 8;
    const float* p0 = xin + (size_t)(kc * 32) * HWn + pix;
#pragma unroll
    for (int h = 0; h < 8; ++h) {
      pa.u[h]     = f2bf(p0[(size_t)(kb + h) * HWn]);
      pa.u[8 + h] = f2bf(p0[(size_t)(16 + kb + h) * HWn]);
    }
    v16bf a = pa.b;
#pragma unroll
    for (int nt = 0; nt < 4; ++nt)
      acc[nt] = wmma_bf16(a, load_b_packed(pkP + (nt * 4 + kc) * 512), acc[nt]);
  }

  int Mb = ((lane >> 4) & 1) * 8;
  bool isS0 = (img % Sn) == 0;
  int bimg = img / Sn;
#pragma unroll
  for (int nt = 0; nt < 4; ++nt) {
    int n = nt * 16 + (lane & 15);
#pragma unroll
    for (int r = 0; r < 8; ++r) {
      float v = acc[nt][r];
      atomicAdd(&csum[n], v);
      atomicAdd(&csum2[n], v * v);
      if (isS0) {
        int p = pix0 + Mb + r;
        x0raw[((size_t)bimg * HWn + p) * 64 + n] = v;
      }
    }
  }
  __syncthreads();
  if (tid < 64) {
    atomicAdd(&chansum[tid], csum[tid]);
    atomicAdd(&chansumsq[tid], csum2[tid]);
    atomicAdd(&ymean_raw[img * 64 + tid], csum[tid]);
  }
}

__global__ void bn_finalize_kernel(const float* __restrict__ chansum,
                                   const float* __restrict__ chansumsq,
                                   const float* __restrict__ gamma,
                                   const float* __restrict__ beta,
                                   float* __restrict__ bn_a, float* __restrict__ bn_b,
                                   const float* __restrict__ ymean_raw,
                                   float* __restrict__ xmean) {
  __shared__ float a_s[64], b_s[64];
  int tid = threadIdx.x;
  if (tid < 64) {
    float cnt = (float)BSn * (float)HWn;
    float mu  = chansum[tid] / cnt;
    float var = chansumsq[tid] / cnt - mu * mu;
    float a = gamma[tid] * rsqrtf(var + 1e-5f);
    float b = beta[tid] - mu * a;
    bn_a[tid] = a; bn_b[tid] = b; a_s[tid] = a; b_s[tid] = b;
  }
  __syncthreads();
  for (int i = tid; i < BSn * 64; i += blockDim.x) {
    int d = i & 63, bs = i >> 6;
    int b_ = bs / Sn, s = bs % Sn;
    xmean[(s * Bn + b_) * 64 + d] =
        a_s[d] * (ymean_raw[bs * 64 + d] * (1.f / (float)HWn)) + b_s[d];
  }
}

__global__ void x0_init_kernel(const float* __restrict__ x0raw,
                               const float* __restrict__ bn_a,
                               const float* __restrict__ bn_b,
                               float* __restrict__ x0,
                               unsigned short* __restrict__ yhat_bf) {
  int i = blockIdx.x * blockDim.x + threadIdx.x;
  if (i >= NPIX * 64) return;
  int d = i & 63;
  float v = bn_a[d] * x0raw[i] + bn_b[d];
  x0[i] = v;
  yhat_bf[i] = f2bf(v);
}

// ------------------------------------------------------- ConvLSTM 3x3 GEMM
// grid: NPIX/16 = 2048 blocks, 256 threads (8 waves); wave w -> n-tiles {2w,2w+1}
// Activations read directly as bf16 (no conversion ALU in hot loop).
__global__ void __launch_bounds__(256)
convlstm_gemm_kernel(const unsigned short* __restrict__ yhat_bf,
                     const unsigned short* __restrict__ hf_bf,
                     const unsigned short* __restrict__ pkC, float* __restrict__ zc) {
  int tile  = blockIdx.x;
  int b     = tile >> 8;             // 256 pixel-tiles per batch image
  int ptile = tile & 255;
  int py  = ptile >> 2;
  int px0 = (ptile & 3) * 16;
  int wave = threadIdx.x >> 5, lane = threadIdx.x & 31;
  int m = lane & 15;

  const unsigned short* actb[2] = { yhat_bf + (size_t)b * HWn * 64,
                                    hf_bf + (size_t)b * HWn * 64 };

  v8f acc0 = {0.f, 0.f, 0.f, 0.f, 0.f, 0.f, 0.f, 0.f};
  v8f acc1 = acc0;
  int nt0 = wave * 2;

#pragma unroll
  for (int tap = 0; tap < 9; ++tap) {
    int dy = tap / 3 - 1, dx = tap % 3 - 1;
    int iy = py + dy;
    int ix = px0 + m + dx;
    bool valid = (iy >= 0) && (iy < Hn) && (ix >= 0) && (ix < Wn);
    int ipix = valid ? (iy * Wn + ix) : 0;
#pragma unroll
    for (int kc = 0; kc < 4; ++kc) {
      const unsigned short* src = actb[kc >> 1] + ((size_t)ipix * 64 + (kc & 1) * 32);
      __builtin_prefetch(src + 64, 0, 1);
      v16bf a = load_a_bf16(src, valid);
      acc0 = wmma_bf16(a, load_b_packed(pkC + (((size_t)nt0 * 9 + tap) * 4 + kc) * 512), acc0);
      acc1 = wmma_bf16(a, load_b_packed(pkC + (((size_t)(nt0 + 1) * 9 + tap) * 4 + kc) * 512), acc1);
    }
  }

  int Mb = ((lane >> 4) & 1) * 8;
#pragma unroll
  for (int r = 0; r < 8; ++r) {
    int p = ptile * 16 + Mb + r;
    size_t row = ((size_t)b * HWn + p) * 256;
    zc[row + wave * 32 + (lane & 15)]      = acc0[r];
    zc[row + wave * 32 + 16 + (lane & 15)] = acc1[r];
  }
}

// gates -> hf (bf16), cf ; accumulate hf spatial sums
__global__ void __launch_bounds__(256)
clstm_pointwise_kernel(const float* __restrict__ zc, const float* __restrict__ clstm_b,
                       unsigned short* __restrict__ hf_bf, float* __restrict__ cf,
                       float* __restrict__ hfmean) {
  __shared__ float hsum[64];
  int tid = threadIdx.x;
  if (tid < 64) hsum[tid] = 0.f;
  __syncthreads();

  int i = blockIdx.x * 256 + tid;       // over NPIX*64
  int d = i & 63;
  size_t zrow = (size_t)(i >> 6) * 256;
  float zi = zc[zrow + d]        + clstm_b[d];
  float zf = zc[zrow + 64 + d]   + clstm_b[64 + d];
  float zo = zc[zrow + 128 + d]  + clstm_b[128 + d];
  float zg = zc[zrow + 192 + d]  + clstm_b[192 + d];
  float c2 = sigm(zf) * cf[i] + sigm(zi) * tanhf(zg);
  float h2 = sigm(zo) * tanhf(c2);
  cf[i] = c2;
  hf_bf[i] = f2bf(h2);
  atomicAdd(&hsum[d], h2);
  __syncthreads();
  int b = blockIdx.x >> 10;             // 1024 blocks per batch
  if (tid < 64) atomicAdd(&hfmean[b * 64 + tid], hsum[tid]);
}

// per-pixel FC 64->64 : y_tick ; accumulate y_tick spatial sums
__global__ void __launch_bounds__(256)
fcf_gemm_kernel(const unsigned short* __restrict__ hf_bf,
                const unsigned short* __restrict__ pkF,
                const float* __restrict__ fcf_b, float* __restrict__ ytick,
                float* __restrict__ ytickmean) {
  __shared__ float ysum[64];
  int tid = threadIdx.x;
  if (tid < 64) ysum[tid] = 0.f;
  __syncthreads();

  int wave = tid >> 5, lane = tid & 31, m = lane & 15;
  size_t pix0 = (size_t)blockIdx.x * 128 + wave * 16;
  const unsigned short* arow = hf_bf + (pix0 + m) * 64;

  v8f z = {0.f, 0.f, 0.f, 0.f, 0.f, 0.f, 0.f, 0.f};
  v8f acc[4];
#pragma unroll
  for (int nt = 0; nt < 4; ++nt) acc[nt] = z;

#pragma unroll
  for (int kc = 0; kc < 2; ++kc) {
    v16bf a = load_a_bf16(arow + kc * 32, true);
#pragma unroll
    for (int nt = 0; nt < 4; ++nt)
      acc[nt] = wmma_bf16(a, load_b_packed(pkF + (nt * 2 + kc) * 512), acc[nt]);
  }

  int Mb = ((lane >> 4) & 1) * 8;
#pragma unroll
  for (int nt = 0; nt < 4; ++nt) {
    int n = nt * 16 + (lane & 15);
#pragma unroll
    for (int r = 0; r < 8; ++r) {
      float v = acc[nt][r] + fcf_b[n];
      ytick[(pix0 + Mb + r) * 64 + n] = v;
      atomicAdd(&ysum[n], v);
    }
  }
  __syncthreads();
  int b = blockIdx.x >> 5;              // 32 blocks per batch
  if (tid < 64) atomicAdd(&ytickmean[b * 64 + tid], ysum[tid]);
}

// ------------------------------------------------ per-step Kalman small chain
__global__ void __launch_bounds__(256)
small_chain_kernel(int t, const float* __restrict__ hfmean,
                   const float* __restrict__ ytickmean, const float* __restrict__ xmean,
                   const float* fcF_w, const float* fcF_b,
                   const float* lq_wih, const float* lq_whh, const float* lq_bih, const float* lq_bhh,
                   const float* lr_wih, const float* lr_whh, const float* lr_bih, const float* lr_bhh,
                   const float* fcQ_w, const float* fcQ_b, const float* fcR_w, const float* fcR_b,
                   float* hQ, float* cQ, float* hR, float* cR,
                   float* P, unsigned short* __restrict__ Kpk) {
  __shared__ float P_s[64 * 64];     // 16 KB
  __shared__ float aug[64 * 128];    // 32 KB
  __shared__ float hfm[64], Fv[64], qin[64], rin[64];
  __shared__ float g_s[256];
  __shared__ float hq_s[64], cq_s[64], hr_s[64], cr_s[64];
  __shared__ float qdiag[64], rdiag[64];
  __shared__ float fcol[64];
  __shared__ float pivinv;

  const int b = blockIdx.x;
  const int tid = threadIdx.x;
  const float invHW = 1.f / (float)HWn;

  if (tid < 64) {
    hfm[tid] = hfmean[b * 64 + tid] * invHW;
    qin[tid] = ytickmean[b * 64 + tid] * invHW;
    rin[tid] = xmean[(t * Bn + b) * 64 + tid];
    hq_s[tid] = hQ[b * 64 + tid];
    cq_s[tid] = cQ[b * 64 + tid];
    hr_s[tid] = hR[b * 64 + tid];
    cr_s[tid] = cR[b * 64 + tid];
  }
  __syncthreads();

  if (tid < 64) {
    float s = fcF_b[tid];
    for (int k = 0; k < 64; ++k) s += hfm[k] * fcF_w[tid * 64 + k];
    Fv[tid] = s;
  }
  // Q-LSTM gates (input = spatial mean of y_tick)
  {
    float s = lq_bih[tid] + lq_bhh[tid];
    for (int k = 0; k < 64; ++k)
      s += qin[k] * lq_wih[tid * 64 + k] + hq_s[k] * lq_whh[tid * 64 + k];
    g_s[tid] = s;
  }
  __syncthreads();
  if (tid < 64) {
    float gi = sigm(g_s[tid]);
    float gf = sigm(g_s[64 + tid]);
    float gg = tanhf(g_s[128 + tid]);
    float go = sigm(g_s[192 + tid]);
    float c2 = gf * cq_s[tid] + gi * gg;
    float h2 = go * tanhf(c2);
    cq_s[tid] = c2; hq_s[tid] = h2;
    cQ[b * 64 + tid] = c2; hQ[b * 64 + tid] = h2;
  }
  __syncthreads();
  // R-LSTM gates (input = spatial mean of xs)
  {
    float s = lr_bih[tid] + lr_bhh[tid];
    for (int k = 0; k < 64; ++k)
      s += rin[k] * lr_wih[tid * 64 + k] + hr_s[k] * lr_whh[tid * 64 + k];
    g_s[tid] = s;
  }
  __syncthreads();
  if (tid < 64) {
    float gi = sigm(g_s[tid]);
    float gf = sigm(g_s[64 + tid]);
    float gg = tanhf(g_s[128 + tid]);
    float go = sigm(g_s[192 + tid]);
    float c2 = gf * cr_s[tid] + gi * gg;
    float h2 = go * tanhf(c2);
    cr_s[tid] = c2; hr_s[tid] = h2;
    cR[b * 64 + tid] = c2; hR[b * 64 + tid] = h2;
  }
  __syncthreads();
  if (tid < 64) {
    float sq = fcQ_b[tid], sr = fcR_b[tid];
    for (int k = 0; k < 64; ++k) {
      sq += hq_s[k] * fcQ_w[tid * 64 + k];
      sr += hr_s[k] * fcR_w[tid * 64 + k];
    }
    qdiag[tid] = expf(sq);
    rdiag[tid] = expf(sr);
  }
  __syncthreads();

  // P = P*FF + Qm ; A = P + Rm (augmented with I)
  for (int e = tid; e < 4096; e += 256) {
    int i = e >> 6, j = e & 63;
    float v = P[(size_t)b * 4096 + e] * Fv[i] * Fv[j] + ((i == j) ? qdiag[i] : 0.f);
    P_s[e] = v;
    aug[i * 128 + j]      = v + ((i == j) ? rdiag[i] : 0.f);
    aug[i * 128 + 64 + j] = (i == j) ? 1.f : 0.f;
  }
  __syncthreads();

  // Gauss-Jordan inversion (SPD-ish, no pivoting)
  for (int kp = 0; kp < 64; ++kp) {
    if (tid == 0) pivinv = 1.f / aug[kp * 128 + kp];
    __syncthreads();
    if (tid < 128) aug[kp * 128 + tid] *= pivinv;
    __syncthreads();
    if (tid < 64) fcol[tid] = aug[tid * 128 + kp];
    __syncthreads();
    for (int e = tid; e < 8192; e += 256) {
      int i = e >> 7, j = e & 127;
      if (i != kp) aug[i * 128 + j] -= fcol[i] * aug[kp * 128 + j];
    }
    __syncthreads();
  }

  // K = P @ Ainv  (store K in left half of aug)
  for (int e = tid; e < 4096; e += 256) {
    int i = e >> 6, j = e & 63;
    float s = 0.f;
    for (int k = 0; k < 64; ++k) s += P_s[i * 64 + k] * aug[k * 128 + 64 + j];
    aug[i * 128 + j] = s;
  }
  __syncthreads();
  // T = (I-K) @ P  (store in right half of aug)
  for (int e = tid; e < 4096; e += 256) {
    int i = e >> 6, j = e & 63;
    float s = 0.f;
    for (int k = 0; k < 64; ++k) {
      float ik = ((i == k) ? 1.f : 0.f) - aug[i * 128 + k];
      s += ik * P_s[k * 64 + j];
    }
    aug[i * 128 + 64 + j] = s;
  }
  __syncthreads();
  // Pnew = T @ (I-K)^T + K diag(r) K^T ; emit P
  for (int e = tid; e < 4096; e += 256) {
    int i = e >> 6, j = e & 63;
    float s = 0.f;
    for (int k = 0; k < 64; ++k) {
      float ikjk = ((j == k) ? 1.f : 0.f) - aug[j * 128 + k];
      s += aug[i * 128 + 64 + k] * ikjk + aug[i * 128 + k] * rdiag[k] * aug[j * 128 + k];
    }
    P[(size_t)b * 4096 + e] = s;
  }
  // emit K pre-packed as bf16 B-fragments (element (K=k,N=n) = K[n][k]):
  // tile tI = nt*2+kc, entry = lane*16+h, n = nt*16+(lane&15),
  // k = kc*32 + ((lane>>4)&1)*8 + h + (h >= 8 ? 8 : 0)
  for (int e = tid; e < 4096; e += 256) {
    int tI = e >> 9;
    int kc = tI & 1, nt = tI >> 1;
    int lane = (e >> 4) & 31, h = e & 15;
    int n = nt * 16 + (lane & 15);
    int k = kc * 32 + ((lane >> 4) & 1) * 8 + h + (h >= 8 ? 8 : 0);
    Kpk[(size_t)b * 4096 + e] = f2bf(aug[n * 128 + k]);
  }
}

// y_hat = y_tick + (x0 - y_tick) @ K^T  ; accumulate feat[b,t,:]
__global__ void __launch_bounds__(256)
yhat_update_kernel(const float* __restrict__ x0, const float* __restrict__ ytick,
                   const unsigned short* __restrict__ Kpk,
                   unsigned short* __restrict__ yhat_bf,
                   float* __restrict__ feat, int t) {
  __shared__ float fsum[64];
  int tid = threadIdx.x;
  if (tid < 64) fsum[tid] = 0.f;
  __syncthreads();

  int b = blockIdx.x >> 5;              // 32 blocks per batch
  int wave = tid >> 5, lane = tid & 31, m = lane & 15;
  size_t pix0 = (size_t)blockIdx.x * 128 + wave * 16;
  const float* xr = x0 + (pix0 + m) * 64;
  const float* yr = ytick + (pix0 + m) * 64;
  const unsigned short* Kp = Kpk + (size_t)b * 4096;

  v8f z = {0.f, 0.f, 0.f, 0.f, 0.f, 0.f, 0.f, 0.f};
  v8f acc[4];
#pragma unroll
  for (int nt = 0; nt < 4; ++nt) acc[nt] = z;

#pragma unroll
  for (int kc = 0; kc < 2; ++kc) {
    BFPack pa;
    int kb = ((lane >> 4) & 1) * 8;
#pragma unroll
    for (int h = 0; h < 8; ++h) {
      int k0 = kc * 32 + kb + h;
      int k1 = kc * 32 + 16 + kb + h;
      pa.u[h]     = f2bf(xr[k0] - yr[k0]);
      pa.u[8 + h] = f2bf(xr[k1] - yr[k1]);
    }
    v16bf a = pa.b;
#pragma unroll
    for (int nt = 0; nt < 4; ++nt)
      acc[nt] = wmma_bf16(a, load_b_packed(Kp + (nt * 2 + kc) * 512), acc[nt]);
  }

  int Mb = ((lane >> 4) & 1) * 8;
#pragma unroll
  for (int nt = 0; nt < 4; ++nt) {
    int n = nt * 16 + (lane & 15);
#pragma unroll
    for (int r = 0; r < 8; ++r) {
      size_t p = pix0 + Mb + r;
      float v = ytick[p * 64 + n] + acc[nt][r];
      yhat_bf[p * 64 + n] = f2bf(v);
      atomicAdd(&fsum[n], v);
    }
  }
  __syncthreads();
  if (tid < 64) atomicAdd(&feat[(b * Sn + t) * 64 + tid], fsum[tid]);
}

// -------------------------------------------------------------- output head
__global__ void head_kernel(const float* __restrict__ feat,
                            const float* fc_w, const float* fc_b,
                            const float* fco_w, const float* fco_b,
                            const float* fca_w, const float* fca_b,
                            float* __restrict__ out) {
  int r = threadIdx.x;                  // row (b*S + s), 0..127
  if (r >= Bn * Sn) return;
  float fm[64];
  for (int d = 0; d < 64; ++d) fm[d] = feat[r * 64 + d] * (1.f / (float)HWn);
  float so = fco_b[0], sa = fca_b[0];
  for (int n = 0; n < 64; ++n) {
    float s = fc_b[n];
    for (int k = 0; k < 64; ++k) s += fm[k] * fc_w[n * 64 + k];
    s = fmaxf(s, 0.f);
    so += s * fco_w[n];
    sa += s * fca_w[n];
  }
  out[r]       = so;   // offset (B,S,1)
  out[128 + r] = sa;   // angle  (B,S,1)
}

// ===========================================================================
extern "C" void kernel_launch(void* const* d_in, const int* in_sizes, int n_in,
                              void* d_out, int out_size, void* d_ws, size_t ws_size,
                              hipStream_t stream) {
  (void)in_sizes; (void)n_in; (void)out_size; (void)ws_size;
  const float* x        = (const float*)d_in[0];
  const float* proj_w   = (const float*)d_in[1];
  const float* bn_gamma = (const float*)d_in[2];
  const float* bn_beta  = (const float*)d_in[3];
  const float* clstm_w  = (const float*)d_in[4];
  const float* clstm_b  = (const float*)d_in[5];
  const float* lq_wih   = (const float*)d_in[6];
  const float* lq_whh   = (const float*)d_in[7];
  const float* lq_bih   = (const float*)d_in[8];
  const float* lq_bhh   = (const float*)d_in[9];
  const float* lr_wih   = (const float*)d_in[10];
  const float* lr_whh   = (const float*)d_in[11];
  const float* lr_bih   = (const float*)d_in[12];
  const float* lr_bhh   = (const float*)d_in[13];
  const float* fcf_w    = (const float*)d_in[14];
  const float* fcf_b    = (const float*)d_in[15];
  const float* fcF_w    = (const float*)d_in[16];
  const float* fcF_b    = (const float*)d_in[17];
  const float* fcQ_w    = (const float*)d_in[18];
  const float* fcQ_b    = (const float*)d_in[19];
  const float* fcR_w    = (const float*)d_in[20];
  const float* fcR_b    = (const float*)d_in[21];
  const float* P_init   = (const float*)d_in[22];
  const float* fc_w     = (const float*)d_in[23];
  const float* fc_b     = (const float*)d_in[24];
  const float* fco_w    = (const float*)d_in[25];
  const float* fco_b    = (const float*)d_in[26];
  const float* fca_w    = (const float*)d_in[27];
  const float* fca_b    = (const float*)d_in[28];
  float* out = (float*)d_out;

  float* W = (float*)d_ws;
  size_t o = 0;
  auto alloc = [&](size_t n) -> float* {
    float* p = W + o;
    o += (n + 63) & ~(size_t)63;
    return p;
  };
  float* x0raw = alloc((size_t)NPIX * 64);
  float* x0    = alloc((size_t)NPIX * 64);
  float* cf    = alloc((size_t)NPIX * 64);
  float* ytick = alloc((size_t)NPIX * 64);
  float* zc    = alloc((size_t)NPIX * 256);
  unsigned short* yhat_bf = (unsigned short*)alloc((size_t)NPIX * 32);  // NPIX*64 u16
  unsigned short* hf_bf   = (unsigned short*)alloc((size_t)NPIX * 32);
  float* Pbuf  = alloc((size_t)Bn * 64 * 64);
  unsigned short* Kpk = (unsigned short*)alloc((size_t)Bn * 4096 / 2);
  float* chansum   = alloc(64);
  float* chansumsq = alloc(64);
  float* bn_a      = alloc(64);
  float* bn_b      = alloc(64);
  float* ymean_raw = alloc((size_t)BSn * 64);
  float* xmean     = alloc((size_t)Sn * Bn * 64);
  float* hfmean    = alloc((size_t)Bn * 64 * 2);     // hfmean + ytickmean contiguous
  float* ytickmean = hfmean + Bn * 64;
  float* lstate    = alloc((size_t)Bn * 64 * 4);     // hQ,cQ,hR,cR
  float* hQ = lstate, *cQ = lstate + 512, *hR = lstate + 1024, *cR = lstate + 1536;
  float* feat = alloc((size_t)Bn * Sn * 64);
  unsigned short* pkC = (unsigned short*)alloc((size_t)16 * 9 * 4 * 512 / 2);
  unsigned short* pkF = (unsigned short*)alloc((size_t)4 * 2 * 512 / 2);
  unsigned short* pkP = (unsigned short*)alloc((size_t)4 * 4 * 512 / 2);

  // ---- init / zero (re-run every call: graph-replay deterministic) ----
  zero_kernel<<<(NPIX * 32 + 255) / 256, 256, 0, stream>>>((float*)hf_bf,
                                                           (long long)NPIX * 32);
  zero_kernel<<<(NPIX * 64 + 255) / 256, 256, 0, stream>>>(cf, (long long)NPIX * 64);
  // chansum..ymean_raw are contiguous (all sizes multiples of 64): 64*4 + 8192
  zero_kernel<<<(8448 + 255) / 256, 256, 0, stream>>>(chansum, 8448);
  zero_kernel<<<4, 256, 0, stream>>>(lstate, 2048);
  zero_kernel<<<(Bn * Sn * 64 + 255) / 256, 256, 0, stream>>>(feat, Bn * Sn * 64);
  init_P_kernel<<<(Bn * 4096 + 255) / 256, 256, 0, stream>>>(P_init, Pbuf);

  // ---- weight packing ----
  {
    int total = 16 * 9 * 4 * 512 + 4 * 2 * 512 + 4 * 4 * 512;
    pack_weights_kernel<<<(total + 255) / 256, 256, 0, stream>>>(
        clstm_w, fcf_w, proj_w, pkC, pkF, pkP);
  }

  // ---- projection + BN stats ----
  proj_gemm_kernel<<<BSn * 32, 256, 0, stream>>>(x, pkP, x0raw, chansum, chansumsq,
                                                 ymean_raw);
  bn_finalize_kernel<<<1, 256, 0, stream>>>(chansum, chansumsq, bn_gamma, bn_beta,
                                            bn_a, bn_b, ymean_raw, xmean);
  x0_init_kernel<<<(NPIX * 64 + 255) / 256, 256, 0, stream>>>(x0raw, bn_a, bn_b, x0,
                                                              yhat_bf);

  // ---- sequence scan ----
  for (int t = 0; t < Sn; ++t) {
    zero_kernel<<<4, 256, 0, stream>>>(hfmean, 1024);
    convlstm_gemm_kernel<<<NPIX / 16, 256, 0, stream>>>(yhat_bf, hf_bf, pkC, zc);
    clstm_pointwise_kernel<<<NPIX * 64 / 256, 256, 0, stream>>>(zc, clstm_b, hf_bf, cf,
                                                                hfmean);
    fcf_gemm_kernel<<<NPIX / 128, 256, 0, stream>>>(hf_bf, pkF, fcf_b, ytick, ytickmean);
    small_chain_kernel<<<Bn, 256, 0, stream>>>(
        t, hfmean, ytickmean, xmean, fcF_w, fcF_b,
        lq_wih, lq_whh, lq_bih, lq_bhh, lr_wih, lr_whh, lr_bih, lr_bhh,
        fcQ_w, fcQ_b, fcR_w, fcR_b, hQ, cQ, hR, cR, Pbuf, Kpk);
    yhat_update_kernel<<<NPIX / 128, 256, 0, stream>>>(x0, ytick, Kpk, yhat_bf, feat, t);
  }

  // ---- heads ----
  head_kernel<<<1, 128, 0, stream>>>(feat, fc_w, fc_b, fco_w, fco_b, fca_w, fca_b, out);
}
